// LocalCRF_11742440587459
// MI455X (gfx1250) — matrix operand order repaired and checked by reading, not compiled, not run on hardware
//
#include <hip/hip_runtime.h>

typedef __attribute__((ext_vector_type(16))) __bf16        v16bf;
typedef __attribute__((ext_vector_type(8)))  float         v8f;
typedef __attribute__((ext_vector_type(8)))  unsigned int  v8u;

// Problem constants (from the reference setup)
#define BB   4
#define CC   150
#define CP   160      // C padded to WMMA tile multiple
#define HH   320
#define WW   320
#define KK   7
#define PAD  3
#define NT   32       // pixels per workgroup tile (2 WMMA N-tiles)
#define PW   (NT + 6) // Q patch width (38)
#define PWP  40       // padded patch row
#define QSTR 166      // LDS B-tile row stride (even elems, odd dword stride -> no bank conflicts)

// ---------------------------------------------------------------------------
// Kernel 0: comp = sigmoid(100 * compatibility), padded to [160 x 160] bf16
// ---------------------------------------------------------------------------
__global__ void comp_sigmoid_kernel(const float* __restrict__ compat,
                                    __bf16* __restrict__ compbf) {
    int idx = blockIdx.x * blockDim.x + threadIdx.x;
    if (idx >= CP * CP) return;
    int o = idx / CP, i = idx % CP;
    float v = 0.0f;
    if (o < CC && i < CC) {
        float c = compat[o * CC + i];
        v = 1.0f / (1.0f + __expf(-100.0f * c));
    }
    compbf[idx] = (__bf16)v;
}

// ---------------------------------------------------------------------------
// Kernel 1: per-pixel channel softmax, Q written as bf16 (online 2-pass)
// ---------------------------------------------------------------------------
__global__ void softmax_kernel(const float* __restrict__ logit,
                               __bf16* __restrict__ q) {
    int idx = blockIdx.x * blockDim.x + threadIdx.x;
    const int HWp = HH * WW;
    if (idx >= BB * HWp) return;
    int b = idx / HWp, pix = idx % HWp;
    const float* base = logit + (size_t)b * CC * HWp + pix;
    float m = -3.4e38f, s = 0.0f;
    for (int c = 0; c < CC; ++c) {
        float x  = base[(size_t)c * HWp];
        float mn = fmaxf(m, x);
        s = s * __expf(m - mn) + __expf(x - mn);
        m = mn;
    }
    float inv = 1.0f / s;
    __bf16* qb = q + (size_t)b * CC * HWp + pix;
    for (int c = 0; c < CC; ++c) {
        float x = base[(size_t)c * HWp];
        qb[(size_t)c * HWp] = (__bf16)(__expf(x - m) * inv);
    }
}

// ---------------------------------------------------------------------------
// Kernel 2: fused 7x7 local CRF filter (VALU, LDS-staged, all 32 lanes busy) +
//           150x150 1x1-conv GEMM via v_wmma_f32_16x16x32_bf16 (10 WMMAs/wave)
//           + out = logit - conv (logit preloaded before WMMAs to hide latency)
// One workgroup: batch b, row i, 32-pixel column tile. 320 threads = 10 waves.
// ---------------------------------------------------------------------------
__global__ __launch_bounds__(320)
void crf_gemm_kernel(const float* __restrict__ Fw,
                     const float* __restrict__ logit,
                     const __bf16* __restrict__ q,
                     const __bf16* __restrict__ compbf,
                     float* __restrict__ out) {
    __shared__ __attribute__((aligned(16))) float  fT[KK * KK][NT];     // filter taps
    __shared__ __attribute__((aligned(16))) __bf16 qpatch[10][KK][PWP]; // per-wave channel patch
    __shared__ __attribute__((aligned(16))) __bf16 qf[NT][QSTR];        // B-tiles: [n=pixel][k=channel]

    const int j0   = blockIdx.x * NT;
    const int i    = blockIdx.y;
    const int b    = blockIdx.z;
    const int tid  = threadIdx.x;
    const int wave = tid >> 5;
    const int lane = tid & 31;

    // --- stage F taps for the 32 pixels: F[b, p, i, j0+j] ---
    for (int e = tid; e < KK * KK * NT; e += 320) {
        int p = e / NT, j = e % NT;
        fT[p][j] = Fw[(((size_t)b * (KK * KK) + p) * HH + i) * WW + (j0 + j)];
    }
    // prefetch the logit rows the epilogue will subtract
    if (tid < 16) {
        __builtin_prefetch(logit + (((size_t)b * CC + tid * 10) * HH + i) * WW + j0, 0, 0);
    }
    __syncthreads();

    // --- 7x7 filter: each wave owns one channel per pass (16 passes cover 160) ---
    const __bf16* qb = q + (size_t)b * CC * HH * WW;
    for (int c0 = 0; c0 < CP; c0 += 10) {
        const int c = c0 + wave;
        // cooperatively load this channel's 7x38 patch (zero-padded)
        for (int e = lane; e < KK * PW; e += 32) {
            int di = e / PW, col = e % PW;
            int r  = i + di - PAD;
            int gc = j0 + col - PAD;
            float v = 0.0f;
            if (c < CC && (unsigned)r < (unsigned)HH && (unsigned)gc < (unsigned)WW)
                v = (float)qb[((size_t)c * HH + r) * WW + gc];
            qpatch[wave][di][col] = (__bf16)v;
        }
        // all 32 lanes compute the 32 filtered outputs for this channel
        {
            float acc = 0.0f;
#pragma unroll
            for (int di = 0; di < KK; ++di)
#pragma unroll
                for (int dj = 0; dj < KK; ++dj)
                    acc += fT[di * KK + dj][lane] * (float)qpatch[wave][di][lane + dj];
            qf[lane][c] = (__bf16)acc;   // zero for c >= CC (patch is zero)
        }
    }
    __syncthreads();

    // --- GEMM: wave w -> output rows [16w, 16w+16), K = 160 in 5 steps of 32,
    //     two N-tiles (columns j0..j0+15 and j0+16..j0+31) sharing the A tile ---
    const int n    = lane & 15;         // output column within N-tile
    const int hi   = lane >> 4;         // lane half selects K sub-range
    const int mrow = wave * 16 + n;     // A-matrix row for this lane (A: M = lane%16)
    const __bf16* arow = compbf + (size_t)mrow * CP;

    // preload epilogue logit values (latency hidden under the WMMAs)
    const float* lrow = logit + ((size_t)b * CC * HH + i) * WW + j0 + n;
    float lg[2][8];
#pragma unroll
    for (int u = 0; u < 2; ++u)
#pragma unroll
        for (int r = 0; r < 8; ++r) {
            int m = wave * 16 + hi * 8 + r;
            lg[u][r] = (m < CC) ? lrow[(size_t)m * HH * WW + u * 16] : 0.0f;
        }

    v8f acc0 = {}, acc1 = {};
#pragma unroll
    for (int t = 0; t < 5; ++t) {
        v8u au, bu0, bu1;
#pragma unroll
        for (int g = 0; g < 8; ++g) {
            // A VGPR layout: v0..3 hold K pairs {hi*8 + 2g}, v4..7 hold {16 + hi*8 + 2(g-4)}
            int ka = 32 * t + ((g < 4) ? (hi * 8 + 2 * g) : (16 + hi * 8 + 2 * (g - 4)));
            au[g] = *(const unsigned int*)(arow + ka);
            // B VGPR layout: same K order as A (v0..3 K pairs {hi*8 + 2g}, v4..7 {16 + hi*8 + 2(g-4)})
            int kb = 32 * t + ((g < 4) ? (hi * 8 + 2 * g) : (16 + hi * 8 + 2 * (g - 4)));
            bu0[g] = *(const unsigned int*)(&qf[n][kb]);
            bu1[g] = *(const unsigned int*)(&qf[16 + n][kb]);
        }
        v16bf a  = __builtin_bit_cast(v16bf, au);
        v16bf b0 = __builtin_bit_cast(v16bf, bu0);
        v16bf b1 = __builtin_bit_cast(v16bf, bu1);
        acc0 = __builtin_amdgcn_wmma_f32_16x16x32_bf16(false, a, false, b0,
                                                       (short)0, acc0, false, false);
        acc1 = __builtin_amdgcn_wmma_f32_16x16x32_bf16(false, a, false, b1,
                                                       (short)0, acc1, false, false);
    }

    // --- epilogue: out = logit - acc.  D layout: lanes 0-15 M=r, lanes 16-31 M=r+8 ---
    float* orow = out + ((size_t)b * CC * HH + i) * WW + j0 + n;
#pragma unroll
    for (int r = 0; r < 8; ++r) {
        int m = wave * 16 + hi * 8 + r;
        if (m < CC) {
            size_t off = (size_t)m * HH * WW;
            orow[off]      = lg[0][r] - acc0[r];
            orow[off + 16] = lg[1][r] - acc1[r];
        }
    }
}

// ---------------------------------------------------------------------------
extern "C" void kernel_launch(void* const* d_in, const int* in_sizes, int n_in,
                              void* d_out, int out_size, void* d_ws, size_t ws_size,
                              hipStream_t stream) {
    (void)in_sizes; (void)n_in; (void)out_size; (void)ws_size;
    const float* Fw     = (const float*)d_in[0];   // [4, 49, 320, 320]
    const float* logit  = (const float*)d_in[1];   // [4, 150, 320, 320]
    const float* compat = (const float*)d_in[2];   // [150, 150, 1, 1]
    float*       out    = (float*)d_out;           // [4, 150, 320, 320]

    char* ws = (char*)d_ws;
    __bf16* compbf = (__bf16*)ws;                  // 160*160*2 = 51,200 B
    __bf16* qbf    = (__bf16*)(ws + 65536);        // 4*150*320*320*2 = 122,880,000 B

    comp_sigmoid_kernel<<<(CP * CP + 255) / 256, 256, 0, stream>>>(compat, compbf);

    const int npix = BB * HH * WW;
    softmax_kernel<<<(npix + 255) / 256, 256, 0, stream>>>(logit, qbf);

    dim3 grid(WW / NT, HH, BB);   // (10, 320, 4)
    crf_gemm_kernel<<<grid, 320, 0, stream>>>(Fw, logit, qbf, compbf, out);
}
